// DiffusionTransformerBlock_67645734912596
// MI455X (gfx1250) — matrix operand order repaired and checked by reading, not compiled, not run on hardware
//
#include <hip/hip_runtime.h>

// ---------------------------------------------------------------------------
// Problem constants (reference: B=2, S=2048, D=2048, H=16, HD=128, HID=5632)
// ---------------------------------------------------------------------------
#define BATCH 2
#define SEQ   2048
#define DIM   2048
#define HEADS 16
#define HDIM  128
#define HIDF  5632

typedef __attribute__((ext_vector_type(16))) __bf16        v16bf;
typedef __attribute__((ext_vector_type(8)))  float         v8f;
typedef __attribute__((ext_vector_type(8)))  unsigned int  v8u;
typedef __attribute__((ext_vector_type(4)))  unsigned int  v4u;
typedef __attribute__((ext_vector_type(4)))  int           v4i;

#ifndef __has_builtin
#define __has_builtin(x) 0
#endif

#if __has_builtin(__builtin_amdgcn_global_load_async_to_lds_b128)
#define HAVE_ASYNC_LDS 1
typedef __attribute__((address_space(1))) v4i* gv4i_p;
typedef __attribute__((address_space(3))) v4i* lv4i_p;
#endif

static __device__ __forceinline__ unsigned short f32_to_bf16(float f) {
  union { float f; unsigned int u; } x; x.f = f;
  unsigned int r = x.u + 0x7FFFu + ((x.u >> 16) & 1u);   // round-to-nearest-even
  return (unsigned short)(r >> 16);
}

static __device__ __forceinline__ v16bf ld_frag(const unsigned int* u) {
  v8u t;
#pragma unroll
  for (int i = 0; i < 8; ++i) t[i] = u[i];
  return __builtin_bit_cast(v16bf, t);
}

static __device__ __forceinline__ v8f wmma_bf16(v16bf a, v16bf b, v8f c) {
  // D = A(16x32 bf16) * B(32x16 bf16) + C(16x16 f32)
  return __builtin_amdgcn_wmma_f32_16x16x32_bf16(false, a, false, b,
                                                 (short)0, c, false, false);
}

// 16-byte global -> LDS copy; async (ASYNCcnt) path on gfx1250 when available.
static __device__ __forceinline__ void copy16_g2l(const unsigned short* g,
                                                  unsigned short* l) {
#ifdef HAVE_ASYNC_LDS
  __builtin_amdgcn_global_load_async_to_lds_b128(
      (gv4i_p)(__attribute__((address_space(1))) void*)(void*)(g),
      (lv4i_p)(__attribute__((address_space(3))) void*)(void*)(l), 0, 0);
#else
  *(v4u*)l = *(const v4u*)g;
#endif
}

static __device__ __forceinline__ void async_wait_all() {
#ifdef HAVE_ASYNC_LDS
#if __has_builtin(__builtin_amdgcn_s_wait_asynccnt)
  __builtin_amdgcn_s_wait_asynccnt(0);
#else
  asm volatile("s_wait_asynccnt 0" ::: "memory");
#endif
#endif
}

// ---------------------------------------------------------------------------
// Weight prep: fp32 [K,N] -> bf16 transposed [N,K] (32x32 LDS-tiled)
// ---------------------------------------------------------------------------
__global__ __launch_bounds__(256) void transpose_to_bf16_kernel(
    const float* __restrict__ W, unsigned short* __restrict__ Wt,
    int Krows, int Ncols)
{
  __shared__ float t[32][33];
  int bx = blockIdx.x * 32;              // n base
  int by = blockIdx.y * 32;              // k base
  int tx = threadIdx.x & 31;
  int ty = threadIdx.x >> 5;             // 0..7
#pragma unroll
  for (int i = 0; i < 32; i += 8)
    t[ty + i][tx] = W[(size_t)(by + ty + i) * Ncols + bx + tx];
  __syncthreads();
#pragma unroll
  for (int i = 0; i < 32; i += 8)
    Wt[(size_t)(bx + ty + i) * Krows + by + tx] = f32_to_bf16(t[tx][ty + i]);
}

// ---------------------------------------------------------------------------
// RMSNorm: fp32 [rows, D] -> bf16, folds norm weight
// ---------------------------------------------------------------------------
__global__ __launch_bounds__(256) void rmsnorm_kernel(
    const float* __restrict__ x, const float* __restrict__ w,
    unsigned short* __restrict__ out, int Dd)
{
  __shared__ float red[8];
  int row = blockIdx.x;
  const float* xr = x + (size_t)row * Dd;
  float ss = 0.f;
  for (int i = threadIdx.x; i < Dd; i += 256) { float f = xr[i]; ss += f * f; }
#pragma unroll
  for (int m = 1; m < 32; m <<= 1) ss += __shfl_xor(ss, m);
  if ((threadIdx.x & 31) == 0) red[threadIdx.x >> 5] = ss;
  __syncthreads();
  float tot = 0.f;
#pragma unroll
  for (int i = 0; i < 8; ++i) tot += red[i];
  float inv = rsqrtf(tot / (float)Dd + 1e-5f);
  for (int i = threadIdx.x; i < Dd; i += 256)
    out[(size_t)row * Dd + i] = f32_to_bf16(xr[i] * inv * w[i]);
}

// ---------------------------------------------------------------------------
// WMMA GEMM: C[M,N] = A[M,K](bf16) x Bt[N,K](bf16, pre-transposed weights)
// Block tile 128x64, K-step 32. 8 waves: 4 (M) x 2 (N); wave = 32 x 32
// (2x2 WMMA accumulators). A/B tiles staged via async global->LDS b128.
// Epilogues: 0=bf16  1=f32  2=RoPE->bf16  3=aux+acc->f32  4=silu(aux)*acc->bf16
// All of M,N,K are multiples of the tile sizes here (no bounds checks).
// ---------------------------------------------------------------------------
__global__ __launch_bounds__(256) void gemm_bf16_kernel(
    const unsigned short* __restrict__ A, const unsigned short* __restrict__ Bt,
    void* __restrict__ Cout, const float* __restrict__ aux,
    const float* __restrict__ cosp, const float* __restrict__ sinp,
    int M, int N, int K, int epi)
{
  __shared__ unsigned short As[128 * 32];  // [row][k]
  __shared__ unsigned short Bs[64 * 32];   // [col][k]

  int tid  = threadIdx.x;
  int wave = tid >> 5;
  int lane = tid & 31;
  int hf   = lane >> 4;
  int idx  = lane & 15;
  int mw   = wave >> 1;     // 0..3
  int nw   = wave & 1;      // 0..1
  int m0   = blockIdx.y * 128;
  int n0   = blockIdx.x * 64;

  v8f acc[2][2];
  v8f zero = {};
  acc[0][0] = zero; acc[0][1] = zero; acc[1][0] = zero; acc[1][1] = zero;

  for (int k0 = 0; k0 < K; k0 += 32) {
    // --- stage A: 128x32 bf16 = 8KB = 512 x 16B, 2 per thread
#pragma unroll
    for (int i = 0; i < 2; ++i) {
      int e = tid + i * 256;
      int r = e >> 2, q = e & 3;
      copy16_g2l(A + (size_t)(m0 + r) * K + k0 + q * 8, As + r * 32 + q * 8);
    }
    // --- stage B: 64x32 bf16 = 4KB = 256 x 16B, 1 per thread
    {
      int r = tid >> 2, q = tid & 3;
      copy16_g2l(Bt + (size_t)(n0 + r) * K + k0 + q * 8, Bs + r * 32 + q * 8);
    }
    async_wait_all();
    __syncthreads();

    unsigned int au[2][8], bu[2][8];
#pragma unroll
    for (int d = 0; d < 8; ++d) {
      int kba = ((d < 4) ? 0 : 16) + hf * 8 + (d & 3) * 2;  // A-frag layout
      int kbb = ((d < 4) ? 0 : 16) + hf * 8 + (d & 3) * 2;  // B-frag layout
#pragma unroll
      for (int t = 0; t < 2; ++t) {
        au[t][d] = *(const unsigned int*)(As + (mw * 32 + t * 16 + idx) * 32 + kba);
        bu[t][d] = *(const unsigned int*)(Bs + (nw * 32 + t * 16 + idx) * 32 + kbb);
      }
    }
    v16bf af0 = ld_frag(au[0]), af1 = ld_frag(au[1]);
    v16bf bf0 = ld_frag(bu[0]), bf1 = ld_frag(bu[1]);
    acc[0][0] = wmma_bf16(af0, bf0, acc[0][0]);
    acc[0][1] = wmma_bf16(af0, bf1, acc[0][1]);
    acc[1][0] = wmma_bf16(af1, bf0, acc[1][0]);
    acc[1][1] = wmma_bf16(af1, bf1, acc[1][1]);
    __syncthreads();
  }

  // --- epilogue ---
#pragma unroll
  for (int mi = 0; mi < 2; ++mi) {
    int r_base = m0 + mw * 32 + mi * 16 + 8 * hf;
#pragma unroll
    for (int v = 0; v < 8; ++v) {
      int gr = r_base + v;
#pragma unroll
      for (int nj = 0; nj < 2; ++nj) {
        float f = acc[mi][nj][v];
        int gc = n0 + nw * 32 + nj * 16 + idx;
        size_t off = (size_t)gr * N + gc;
        if (epi == 0) {
          ((unsigned short*)Cout)[off] = f32_to_bf16(f);
        } else if (epi == 1) {
          ((float*)Cout)[off] = f;
        } else if (epi == 2) {
          // RoPE: pair partner lives in adjacent lane (col parity == lane parity)
          float p = __shfl_xor(f, 1);
          int s = gr & (SEQ - 1);
          int pr = (gc & (HDIM - 1)) >> 1;
          float cs = cosp[s * (HDIM / 2) + pr];
          float sn = sinp[s * (HDIM / 2) + pr];
          float res = f * cs + (((gc & 1) ? p : -p) * sn);
          ((unsigned short*)Cout)[off] = f32_to_bf16(res);
        } else if (epi == 3) {
          ((float*)Cout)[off] = aux[off] + f;
        } else {
          float g = aux[off];
          float sg = g / (1.0f + __expf(-g));      // silu
          ((unsigned short*)Cout)[off] = f32_to_bf16(sg * f);
        }
      }
    }
  }
}

// ---------------------------------------------------------------------------
// Flash attention (no mask): block = 128 q-rows of one (b,h); 8 waves x 16 rows.
// K tile async-staged to LDS; V tile transposed through VGPRs; running softmax.
// ---------------------------------------------------------------------------
__global__ __launch_bounds__(256) void flash_attn_kernel(
    const unsigned short* __restrict__ Qm, const unsigned short* __restrict__ Km,
    const unsigned short* __restrict__ Vm, unsigned short* __restrict__ Om)
{
  __shared__ unsigned short Ks[32 * 128];      // [kk][d]
  __shared__ unsigned short Vt[128 * 32];      // [d][kk]  (transposed)
  __shared__ unsigned short Ps[8 * 16 * 32];   // per-wave P tile [16][32]

  int tid  = threadIdx.x;
  int wave = tid >> 5;
  int lane = tid & 31;
  int hf   = lane >> 4;
  int idx  = lane & 15;
  int h = blockIdx.y, b = blockIdx.z;
  int q0 = blockIdx.x * 128 + wave * 16;

  // Preload Q fragments for all 4 K-chunks of HD=128
  unsigned int qf[4][8];
  {
    const unsigned short* qrow =
        Qm + ((size_t)(b * SEQ + q0 + idx)) * DIM + h * HDIM;
#pragma unroll
    for (int kc = 0; kc < 4; ++kc)
#pragma unroll
      for (int d = 0; d < 8; ++d) {
        int kb = ((d < 4) ? 0 : 16) + hf * 8 + (d & 3) * 2;
        qf[kc][d] = *(const unsigned int*)(qrow + kc * 32 + kb);
      }
  }

  v8f acc[8];
#pragma unroll
  for (int i = 0; i < 8; ++i) acc[i] = (v8f){};
  float rmax[8], rsum[8];
#pragma unroll
  for (int v = 0; v < 8; ++v) { rmax[v] = -1e30f; rsum[v] = 0.f; }
  const float scale = 0.0883883476483184f;   // 1/sqrt(128)

  for (int kk = 0; kk < SEQ; kk += 32) {
    // --- stage K tile (row-major) via async copies: 8KB = 512 x 16B
#pragma unroll
    for (int i = 0; i < 2; ++i) {
      int e = tid + i * 256;
      int r = e >> 4, q = e & 15;
      copy16_g2l(Km + ((size_t)(b * SEQ + kk + r)) * DIM + h * HDIM + q * 8,
                 Ks + r * 128 + q * 8);
    }
    // --- stage V transposed through VGPRs
#pragma unroll
    for (int i = 0; i < 8; ++i) {
      int dw = i * 256 + tid;         // 0..2047
      int r  = dw >> 6;               // 0..31
      int c2 = (dw & 63) << 1;        // 0..126 even
      unsigned int vv = *(const unsigned int*)(
          Vm + ((size_t)(b * SEQ + kk + r)) * DIM + h * HDIM + c2);
      Vt[(c2)     * 32 + r] = (unsigned short)(vv & 0xffffu);
      Vt[(c2 + 1) * 32 + r] = (unsigned short)(vv >> 16);
    }
    async_wait_all();
    __syncthreads();

    // --- scores: S[16 x 32] = Q(16x128) . K^T
    v8f sc[2];
#pragma unroll
    for (int j = 0; j < 2; ++j) {
      v8f s = (v8f){};
#pragma unroll
      for (int kc = 0; kc < 4; ++kc) {
        unsigned int bu[8];
#pragma unroll
        for (int d = 0; d < 8; ++d) {
          int kb = ((d < 4) ? 0 : 16) + hf * 8 + (d & 3) * 2;
          bu[d] = *(const unsigned int*)(Ks + (j * 16 + idx) * 128 + kc * 32 + kb);
        }
        s = wmma_bf16(ld_frag(qf[kc]), ld_frag(bu), s);
      }
      sc[j] = s;
    }

    // --- online softmax + write P tile to per-wave LDS region
#pragma unroll
    for (int v = 0; v < 8; ++v) {
      float s0 = sc[0][v] * scale, s1 = sc[1][v] * scale;
      float tm = fmaxf(s0, s1);
      tm = fmaxf(tm, __shfl_xor(tm, 1));
      tm = fmaxf(tm, __shfl_xor(tm, 2));
      tm = fmaxf(tm, __shfl_xor(tm, 4));
      tm = fmaxf(tm, __shfl_xor(tm, 8));
      float nm = fmaxf(rmax[v], tm);
      float r  = __expf(rmax[v] - nm);
      rmax[v] = nm;
      float p0 = __expf(s0 - nm), p1 = __expf(s1 - nm);
      float ps = p0 + p1;
      ps += __shfl_xor(ps, 1);
      ps += __shfl_xor(ps, 2);
      ps += __shfl_xor(ps, 4);
      ps += __shfl_xor(ps, 8);
      rsum[v] = rsum[v] * r + ps;
#pragma unroll
      for (int dt = 0; dt < 8; ++dt) acc[dt][v] *= r;
      int m = v + 8 * hf;
      Ps[wave * 512 + m * 32 + idx]      = f32_to_bf16(p0);
      Ps[wave * 512 + m * 32 + 16 + idx] = f32_to_bf16(p1);
    }
    asm volatile("s_wait_dscnt 0" ::: "memory");

    // --- re-pack P as A fragment; accumulate P . V
    unsigned int pf[8];
#pragma unroll
    for (int d = 0; d < 8; ++d) {
      int kb = ((d < 4) ? 0 : 16) + hf * 8 + (d & 3) * 2;
      pf[d] = *(const unsigned int*)(Ps + wave * 512 + idx * 32 + kb);
    }
    v16bf pfr = ld_frag(pf);
#pragma unroll
    for (int dt = 0; dt < 8; ++dt) {
      unsigned int bu[8];
#pragma unroll
      for (int d = 0; d < 8; ++d) {
        int kb = ((d < 4) ? 0 : 16) + hf * 8 + (d & 3) * 2;
        bu[d] = *(const unsigned int*)(Vt + (dt * 16 + idx) * 32 + kb);
      }
      acc[dt] = wmma_bf16(pfr, ld_frag(bu), acc[dt]);
    }
    __syncthreads();
  }

  // --- normalize and write O (bf16, [b*S+s, h*HD+d] layout)
#pragma unroll
  for (int v = 0; v < 8; ++v) {
    int m = v + 8 * hf;
    float inv = 1.0f / rsum[v];
    size_t row = (size_t)(b * SEQ + q0 + m);
#pragma unroll
    for (int dt = 0; dt < 8; ++dt)
      Om[row * DIM + h * HDIM + dt * 16 + idx] = f32_to_bf16(acc[dt][v] * inv);
  }
}

// ---------------------------------------------------------------------------
// Host orchestration
// ---------------------------------------------------------------------------
extern "C" void kernel_launch(void* const* d_in, const int* in_sizes, int n_in,
                              void* d_out, int out_size, void* d_ws, size_t ws_size,
                              hipStream_t stream) {
  (void)in_sizes; (void)n_in; (void)out_size; (void)ws_size;
  const int M = BATCH * SEQ;     // 4096

  const float* x    = (const float*)d_in[0];
  const float* cosp = (const float*)d_in[1];
  const float* sinp = (const float*)d_in[2];
  const float* wq   = (const float*)d_in[3];
  const float* wk   = (const float*)d_in[4];
  const float* wv   = (const float*)d_in[5];
  const float* wo   = (const float*)d_in[6];
  const float* w1   = (const float*)d_in[7];
  const float* w2   = (const float*)d_in[8];
  const float* w3   = (const float*)d_in[9];
  const float* anw  = (const float*)d_in[10];
  const float* fnw  = (const float*)d_in[11];
  float* out = (float*)d_out;

  char* p = (char*)d_ws;
  auto alloc = [&](size_t bytes) {
    void* r = (void*)p;
    p += (bytes + 255) & ~(size_t)255;
    return r;
  };
  // activations
  unsigned short* xn = (unsigned short*)alloc((size_t)M * DIM * 2);
  unsigned short* qb = (unsigned short*)alloc((size_t)M * DIM * 2);
  unsigned short* kb = (unsigned short*)alloc((size_t)M * DIM * 2);
  unsigned short* vb = (unsigned short*)alloc((size_t)M * DIM * 2);
  unsigned short* ao = (unsigned short*)alloc((size_t)M * DIM * 2);
  float*          hb = (float*)alloc((size_t)M * DIM * 4);
  unsigned short* hn = (unsigned short*)alloc((size_t)M * DIM * 2);
  float*          g1 = (float*)alloc((size_t)M * HIDF * 4);
  unsigned short* ub = (unsigned short*)alloc((size_t)M * HIDF * 2);
  // pre-transposed bf16 weights [N,K]
  unsigned short* wtq = (unsigned short*)alloc((size_t)DIM * DIM * 2);
  unsigned short* wtk = (unsigned short*)alloc((size_t)DIM * DIM * 2);
  unsigned short* wtv = (unsigned short*)alloc((size_t)DIM * DIM * 2);
  unsigned short* wto = (unsigned short*)alloc((size_t)DIM * DIM * 2);
  unsigned short* wt1 = (unsigned short*)alloc((size_t)HIDF * DIM * 2);
  unsigned short* wt3 = (unsigned short*)alloc((size_t)HIDF * DIM * 2);
  unsigned short* wt2 = (unsigned short*)alloc((size_t)DIM * HIDF * 2);

  dim3 blk(256);
  dim3 tDD(DIM / 32, DIM / 32);
  dim3 tDH(HIDF / 32, DIM / 32);   // W [D,HID]
  dim3 tHD(DIM / 32, HIDF / 32);   // W [HID,D]
  dim3 gD(DIM / 64, M / 128);      // N=2048 GEMMs
  dim3 gH(HIDF / 64, M / 128);     // N=5632 GEMMs

  // 0. weight prep: f32 [K,N] -> bf16 [N,K]
  transpose_to_bf16_kernel<<<tDD, blk, 0, stream>>>(wq, wtq, DIM, DIM);
  transpose_to_bf16_kernel<<<tDD, blk, 0, stream>>>(wk, wtk, DIM, DIM);
  transpose_to_bf16_kernel<<<tDD, blk, 0, stream>>>(wv, wtv, DIM, DIM);
  transpose_to_bf16_kernel<<<tDD, blk, 0, stream>>>(wo, wto, DIM, DIM);
  transpose_to_bf16_kernel<<<tDH, blk, 0, stream>>>(w1, wt1, DIM, HIDF);
  transpose_to_bf16_kernel<<<tDH, blk, 0, stream>>>(w3, wt3, DIM, HIDF);
  transpose_to_bf16_kernel<<<tHD, blk, 0, stream>>>(w2, wt2, HIDF, DIM);

  // 1. xn = rmsnorm(x) (bf16)
  rmsnorm_kernel<<<M, blk, 0, stream>>>(x, anw, xn, DIM);
  // 2. q/k (RoPE fused), v
  gemm_bf16_kernel<<<gD, blk, 0, stream>>>(xn, wtq, qb, nullptr, cosp, sinp, M, DIM, DIM, 2);
  gemm_bf16_kernel<<<gD, blk, 0, stream>>>(xn, wtk, kb, nullptr, cosp, sinp, M, DIM, DIM, 2);
  gemm_bf16_kernel<<<gD, blk, 0, stream>>>(xn, wtv, vb, nullptr, nullptr, nullptr, M, DIM, DIM, 0);
  // 3. attention
  flash_attn_kernel<<<dim3(SEQ / 128, HEADS, BATCH), blk, 0, stream>>>(qb, kb, vb, ao);
  // 4. h = x + attn @ wo (f32)
  gemm_bf16_kernel<<<gD, blk, 0, stream>>>(ao, wto, hb, x, nullptr, nullptr, M, DIM, DIM, 3);
  // 5. hn = rmsnorm(h) (bf16)
  rmsnorm_kernel<<<M, blk, 0, stream>>>(hb, fnw, hn, DIM);
  // 6. g1 = hn @ w1 (f32); u = silu(g1) * (hn @ w3) (bf16)
  gemm_bf16_kernel<<<gH, blk, 0, stream>>>(hn, wt1, g1, nullptr, nullptr, nullptr, M, HIDF, DIM, 1);
  gemm_bf16_kernel<<<gH, blk, 0, stream>>>(hn, wt3, ub, g1, nullptr, nullptr, M, HIDF, DIM, 4);
  // 7. out = h + u @ w2 (f32)
  gemm_bf16_kernel<<<gD, blk, 0, stream>>>(ub, wt2, out, hb, nullptr, nullptr, M, DIM, HIDF, 3);
}
